// AttentiveTransformer_49297634624103
// MI455X (gfx1250) — matrix operand order repaired and checked by reading, not compiled, not run on hardware
//
#include <hip/hip_runtime.h>
#include <math.h>

typedef float    v8f  __attribute__((ext_vector_type(8)));
typedef _Float16 v16h __attribute__((ext_vector_type(16)));
typedef _Float16 v8h  __attribute__((ext_vector_type(8)));
typedef _Float16 v4h  __attribute__((ext_vector_type(4)));

#define BCHUNK 128      // ghost batch (rows per workgroup)
#define DG 512          // output features
#define DIN 256         // input features (K)
#define LDS_STRIDE 516  // xs row stride: 512 + 4 pad
#define AST 264         // A-panel LDS row stride in halves (132 dw, %64==4 -> conflict-free frags)
#define EPSV 1e-5f
#define NTHREADS 512

union AB16 { v16h v; v8h h[2]; };

// ---- prologue: split W into f16 hi/lo once (reused by all 512 workgroups) ----
__global__ void wsplit_kernel(const float* __restrict__ W,
                              _Float16* __restrict__ Whi,
                              _Float16* __restrict__ Wlo, int n)
{
    int i = blockIdx.x * blockDim.x + threadIdx.x;
    if (i < n) {
        float x = W[i];
        _Float16 h = (_Float16)x;
        Whi[i] = h;
        Wlo[i] = (_Float16)(x - (float)h);
    }
}

__global__ __launch_bounds__(NTHREADS, 1)
void attentive_fused_kernel(const float* __restrict__ priors,
                            const float* __restrict__ feat,
                            const _Float16* __restrict__ Whi,
                            const _Float16* __restrict__ Wlo,
                            const float* __restrict__ gamma,
                            const float* __restrict__ beta,
                            float* __restrict__ out)
{
    extern __shared__ float lds[];
    float* xs  = lds;                              // [128][516], live AFTER the K-loop
    float* s_a = lds + BCHUNK * LDS_STRIDE;        // [512] per-column scale
    float* s_b = s_a + DG;                         // [512] per-column shift
    // A hi/lo panels alias xs: live only DURING the K-loop (barrier separates)
    _Float16* Ahi = (_Float16*)lds;                // [128][264]
    _Float16* Alo = Ahi + BCHUNK * AST;            // [128][264]

    const int tid  = threadIdx.x;
    const int lane = tid & 31;
    const int wave = tid >> 5;
    const int g    = blockIdx.x;                   // chunk index
    const int row_base = g * BCHUNK;

    // ======== stage A chunk: coalesced load + one-time hi/lo split into LDS ====
    {
        const float* src = feat + (size_t)row_base * DIN;
        for (int idx = tid; idx < BCHUNK * DIN / 4; idx += NTHREADS) {
            int r  = idx >> 6;                     // DIN/4 = 64 float4 per row
            int c4 = (idx & 63) * 4;
            float4 x = *(const float4*)(src + (size_t)r * DIN + c4);
            _Float16 h0 = (_Float16)x.x, h1 = (_Float16)x.y;
            _Float16 h2 = (_Float16)x.z, h3 = (_Float16)x.w;
            v4h hi = {h0, h1, h2, h3};
            v4h lo = {(_Float16)(x.x - (float)h0), (_Float16)(x.y - (float)h1),
                      (_Float16)(x.z - (float)h2), (_Float16)(x.w - (float)h3)};
            *(v4h*)(Ahi + r * AST + c4) = hi;
            *(v4h*)(Alo + r * AST + c4) = lo;
        }
    }
    __syncthreads();

    // ================= GEMM: acc = feat[chunk] @ W^T  (f16 split precision) ==
    // 16 waves as 2x8 grid of 64x64 blocks; 4x4 WMMA tiles per wave.
    // acc += ah*bh + al*bh + ah*bl   (lo*lo term ~2^-22, dropped)
    const int m0 = (wave >> 3) * 64;               // 0 or 64
    const int n0 = (wave & 7) * 64;                // 0..448
    const int fr = lane & 15;                      // fragment row (M for A, N for B)
    const int acho = (lane >> 4) * 8;              // A fragment K sub-offset
    const int bko  = (lane >> 4) * 8;              // B fragment K sub-offset

    v8f acc[4][4] = {};
    const _Float16* Afrag = Ahi + (m0 + fr) * AST + acho;   // Alo at fixed delta
    const ptrdiff_t AHL   = (ptrdiff_t)(Alo - Ahi);
    const _Float16* Bhib  = Whi + (size_t)(n0 + fr) * DIN + bko;
    const ptrdiff_t BHL   = (ptrdiff_t)(Wlo - Whi);

    for (int k0 = 0; k0 < DIN; k0 += 32) {
        AB16 ah[4], al[4];
#pragma unroll
        for (int i = 0; i < 4; ++i) {
            const _Float16* pa = Afrag + i * 16 * AST + k0;
            ah[i].h[0] = *(const v8h*)(pa);
            ah[i].h[1] = *(const v8h*)(pa + 16);
            al[i].h[0] = *(const v8h*)(pa + AHL);
            al[i].h[1] = *(const v8h*)(pa + AHL + 16);
        }
#pragma unroll
        for (int j = 0; j < 4; ++j) {
            const _Float16* qb = Bhib + (size_t)j * 16 * DIN + k0;
            AB16 bh, bl;
            bh.h[0] = *(const v8h*)(qb);
            bh.h[1] = *(const v8h*)(qb + 16);
            bl.h[0] = *(const v8h*)(qb + BHL);
            bl.h[1] = *(const v8h*)(qb + BHL + 16);
#pragma unroll
            for (int i = 0; i < 4; ++i) {
                acc[i][j] = __builtin_amdgcn_wmma_f32_16x16x32_f16(
                    false, ah[i].v, false, bh.v, (short)0, acc[i][j], false, false);
                acc[i][j] = __builtin_amdgcn_wmma_f32_16x16x32_f16(
                    false, al[i].v, false, bh.v, (short)0, acc[i][j], false, false);
                acc[i][j] = __builtin_amdgcn_wmma_f32_16x16x32_f16(
                    false, ah[i].v, false, bl.v, (short)0, acc[i][j], false, false);
            }
        }
    }
    __syncthreads();   // A panels dead; xs (aliased) becomes live

    // C layout: VGPR e -> lanes 0-15: (M=e, N=lane); lanes 16-31: (M=e+8, N=lane-16)
    const int cm = (lane < 16) ? 0 : 8;
#pragma unroll
    for (int i = 0; i < 4; ++i)
#pragma unroll
        for (int j = 0; j < 4; ++j)
#pragma unroll
            for (int e = 0; e < 8; ++e)
                xs[(m0 + i * 16 + cm + e) * LDS_STRIDE + (n0 + j * 16 + (lane & 15))] =
                    acc[i][j][e];
    __syncthreads();

    // ================= Ghost BatchNorm stats (per column over 128 rows) =====
    {
        float s = 0.f, ss = 0.f;
        const float* colp = xs + tid;
#pragma unroll 4
        for (int r = 0; r < BCHUNK; ++r) {
            float v = colp[r * LDS_STRIDE];
            s += v; ss += v * v;
        }
        float mean = s * (1.0f / BCHUNK);
        float var  = ss * (1.0f / BCHUNK) - mean * mean;
        float rstd = rsqrtf(var + EPSV);
        float ga = gamma[tid];
        float sc = ga * rstd;
        s_a[tid] = sc;
        s_b[tid] = beta[tid] - mean * sc;
    }
    __syncthreads();

    // ================= normalize + multiply priors =================
    {
        const float* pr = priors + (size_t)row_base * DG;
        float sc = s_a[tid], sh = s_b[tid];
        for (int r = 0; r < BCHUNK; ++r) {
            float v = xs[r * LDS_STRIDE + tid];
            v = v * sc + sh;
            v *= pr[(size_t)r * DG + tid];
            xs[r * LDS_STRIDE + tid] = v;
        }
    }
    __syncthreads();

    // ================= sparsemax per row (wave per row) =================
    for (int r = wave; r < BCHUNK; r += 16) {
        float z[16];
        float zmax = -INFINITY;
#pragma unroll
        for (int e = 0; e < 16; ++e) {
            z[e] = xs[r * LDS_STRIDE + lane + 32 * e];
            zmax = fmaxf(zmax, z[e]);
        }
#pragma unroll
        for (int off = 16; off >= 1; off >>= 1)
            zmax = fmaxf(zmax, __shfl_xor(zmax, off, 32));
#pragma unroll
        for (int e = 0; e < 16; ++e) z[e] -= zmax;

        // Michelot fixed point: tau = (sum_{z>tau} z - 1) / |{z>tau}|
        float sum = 0.f;
#pragma unroll
        for (int e = 0; e < 16; ++e) sum += z[e];
#pragma unroll
        for (int off = 16; off >= 1; off >>= 1)
            sum += __shfl_xor(sum, off, 32);
        float kf  = (float)DG;
        float tau = (sum - 1.0f) * (1.0f / kf);

        for (int it = 0; it < 64; ++it) {
            float ns = 0.f, nk = 0.f;
#pragma unroll
            for (int e = 0; e < 16; ++e) {
                if (z[e] > tau) { ns += z[e]; nk += 1.0f; }
            }
#pragma unroll
            for (int off = 16; off >= 1; off >>= 1) {
                ns += __shfl_xor(ns, off, 32);
                nk += __shfl_xor(nk, off, 32);
            }
            if (nk == kf) break;          // support stable -> tau is exact
            kf  = nk;
            tau = (ns - 1.0f) / nk;
        }

        float* op = out + (size_t)(row_base + r) * DG;
#pragma unroll
        for (int e = 0; e < 16; ++e) {
            float v = z[e] - tau;
            op[lane + 32 * e] = v > 0.f ? v : 0.f;
        }
    }
}

extern "C" void kernel_launch(void* const* d_in, const int* in_sizes, int n_in,
                              void* d_out, int out_size, void* d_ws, size_t ws_size,
                              hipStream_t stream) {
    const float* priors = (const float*)d_in[0];
    const float* feat   = (const float*)d_in[1];
    const float* W      = (const float*)d_in[2];
    const float* gamma  = (const float*)d_in[3];
    const float* beta   = (const float*)d_in[4];
    float* out = (float*)d_out;

    const int B = in_sizes[1] / DIN;       // 65536
    const int nchunk = B / BCHUNK;         // 512 workgroups, one per ghost batch
    const int wn = DG * DIN;               // 131072 W elements

    _Float16* Whi = (_Float16*)d_ws;
    _Float16* Wlo = Whi + wn;

    wsplit_kernel<<<(wn + 255) / 256, 256, 0, stream>>>(W, Whi, Wlo, wn);

    const size_t shmem = (size_t)(BCHUNK * LDS_STRIDE + 2 * DG) * sizeof(float);
    attentive_fused_kernel<<<nchunk, NTHREADS, shmem, stream>>>(
        priors, feat, Whi, Wlo, gamma, beta, out);
}
